// SelfAttention_17824114279099
// MI455X (gfx1250) — matrix operand order repaired and checked by reading, not compiled, not run on hardware
//
#include <hip/hip_runtime.h>

// ============================================================================
// Causal self-attention with RoPE for MI455X (gfx1250, wave32, WMMA).
// Pass 1: RoPE + (log2e/sqrt(D)) scale + f32->f16 pack of Q,K,V into ws.
//         (needs ws_size >= 3 * B*S*128 * 2 bytes ~= 25.2 MB)
// Pass 2: flash attention, v_wmma_f32_16x16x32_f16 for QK^T and P*V.
//         Row sums via P*ones WMMA; row max via packed-f16 shuffle reduce.
//         Wave-uniform tile classification: skip fully-masked tiles, apply
//         the causal mask only on the single diagonal tile per wave.
// ============================================================================

typedef __attribute__((ext_vector_type(16))) _Float16 v16h;
typedef __attribute__((ext_vector_type(2)))  _Float16 v2h;
typedef __attribute__((ext_vector_type(8)))  float    v8f;

#define HEAD_DIM 128
#define SEQ      4096
#define QTILE    128         // query rows per workgroup (16 per wave, 8 waves)
#define KTILE    32          // keys per inner tile
#define NWAVES   8
// mask and scale in log2 domain: softmax(s) == 2^(s*log2e) normalized
#define MASK2    (-14426.950408889634f)      // -10000 * log2(e)
#define QSCALE   (0.12752008805648452f)      // log2(e) / sqrt(128)

union HFrag { v16h h; uint4 u[2]; };
union PKU   { v2h h; float f; unsigned u; };

// ---------------------------------------------------------------------------
// Pass 1: one thread per (b, s, matrix, even/odd pair).
// ---------------------------------------------------------------------------
__global__ __launch_bounds__(256) void rope_pack_kernel(
    const float* __restrict__ qkv,
    _Float16* __restrict__ Qh, _Float16* __restrict__ Kh,
    _Float16* __restrict__ Vh, int BS)
{
  int idx = blockIdx.x * blockDim.x + threadIdx.x;
  if (idx >= BS * 192) return;
  int i  = idx & 63;            // freq-pair index 0..63 (dims 2i, 2i+1)
  int m  = (idx >> 6) % 3;      // 0=q, 1=k, 2=v
  int bs = idx / 192;           // b*S + s
  int s  = bs % SEQ;            // sequence position

  const float2 x = ((const float2*)qkv)[(size_t)bs * 192 + m * 64 + i];
  float oe, oo;
  if (m == 2) {                 // V: straight copy
    oe = x.x; oo = x.y;
  } else {                      // Q, K: rotate
    float inv_freq = __expf(-(float)i * (9.210340371976184f / 64.0f));
    float ang = (float)s * inv_freq;
    float sn, cs;
    sincosf(ang, &sn, &cs);     // accurate path: ang up to ~4095 rad
    oe = x.x * cs - x.y * sn;
    oo = x.x * sn + x.y * cs;
    if (m == 0) { oe *= QSCALE; oo *= QSCALE; }  // fold log2e/sqrt(D) into Q
  }
  union { _Float16 h[2]; unsigned u; } p;
  p.h[0] = (_Float16)oe; p.h[1] = (_Float16)oo;
  _Float16* dst = (m == 0) ? Qh : (m == 1) ? Kh : Vh;
  ((unsigned*)dst)[(size_t)bs * 64 + i] = p.u;
}

// ---------------------------------------------------------------------------
// Pass 2: flash attention, double-buffered tiles, 8 waves/block.
// ---------------------------------------------------------------------------
__global__ __launch_bounds__(256) void fattn_kernel(
    const _Float16* __restrict__ Qh, const _Float16* __restrict__ Kh,
    const _Float16* __restrict__ Vh, float* __restrict__ out, int S)
{
  // K tiles row-major [key][dim], pad 136 -> conflict-free b128 row reads
  __shared__ alignas(16) _Float16 Ks[2][KTILE][HEAD_DIM + 8];
  // V tiles transposed [dim][key], pad 40
  __shared__ alignas(16) _Float16 Vt[2][HEAD_DIM][KTILE + 8];
  // per-wave P staging (C-layout -> A-layout round trip)
  __shared__ alignas(16) _Float16 Ps[NWAVES][16][KTILE + 8];

  const int QT   = S / QTILE;
  const int b    = blockIdx.x / QT;
  const int qt   = blockIdx.x % QT;
  const int tid  = threadIdx.x;
  const int wave = tid >> 5;
  const int lane = tid & 31;
  const int n    = lane & 15;     // column index within 16-wide frags
  const int g    = lane >> 4;     // half-wave group
  const int q0   = qt * QTILE + wave * 16;
  const size_t bbase = (size_t)b * S;

  // cooperative-load geometry: 256 uint4 per tile, 1 per thread
  const int lr = tid >> 3;              // key row 0..31
  const int lc = tid & 7;               // uint4 column (8 halves each)

  auto loadKV = [&](int k0, uint4& kd, uint4& vd) {
    kd = ((const uint4*)(Kh + (bbase + k0 + lr) * HEAD_DIM))[lc];
    vd = ((const uint4*)(Vh + (bbase + k0 + lr) * HEAD_DIM))[lc];
  };
  auto storeKV = [&](int buf, uint4 kd, uint4 vd) {
    *(uint4*)&Ks[buf][lr][lc * 8] = kd;
    union { uint4 u; _Float16 h[8]; } t;
    t.u = vd;
    #pragma unroll
    for (int j = 0; j < 8; ++j)         // transpose V into Vt[dim][key]
      Vt[buf][lc * 8 + j][lr] = t.h[j];
  };

  // ---- Q A-fragments: lane holds row q0+n; two b128 per 32-wide K chunk ----
  HFrag a[4];
  {
    const uint4* qp = (const uint4*)(Qh + (bbase + q0 + n) * HEAD_DIM);
    #pragma unroll
    for (int c = 0; c < 4; ++c) {
      a[c].u[0] = qp[c * 4 + 2 * g];     // dims c*32 + g*16     (A halves 0..7)
      a[c].u[1] = qp[c * 4 + 2 * g + 1]; // dims c*32 + g*16 + 8 (A halves 8..15)
    }
  }

  // all-ones B fragment: P * ones accumulates the softmax denominator
  v16h ones;
  #pragma unroll
  for (int j = 0; j < 16; ++j) ones[j] = (_Float16)1.0f;

  v8f o[8], ol = {};
  #pragma unroll
  for (int c = 0; c < 8; ++c) o[c] = {};
  float m_i[8];
  #pragma unroll
  for (int r = 0; r < 8; ++r) m_i[r] = -1e30f;

  const int nt = ((qt + 1) * QTILE) / KTILE;   // causal KV extent for this WG

  // ---- prologue: stage tile 0 into buffer 0 ----
  uint4 kd, vd;
  loadKV(0, kd, vd);
  storeKV(0, kd, vd);
  __syncthreads();

  for (int kt = 0; kt < nt; ++kt) {
    const int  k0      = kt * KTILE;
    const int  buf     = kt & 1;
    const bool hasNext = (kt + 1 < nt);

    // issue next tile's global loads NOW; latency hides behind the WMMAs
    if (hasNext) loadKV(k0 + KTILE, kd, vd);
    if (kt + 2 < nt && tid < 32) {     // warm L2/L0 for the tile after next
      __builtin_prefetch(Kh + (bbase + k0 + 2 * KTILE + tid) * HEAD_DIM, 0, 1);
      __builtin_prefetch(Vh + (bbase + k0 + 2 * KTILE + tid) * HEAD_DIM, 0, 1);
    }

    // wave-uniform: this tile is entirely above the diagonal for this wave;
    // every p underflows to exactly 0 in f32, so skip all compute.
    if (k0 <= q0 + 15) {
      // ---- scores: batch all B-frag LDS loads, then back-to-back WMMAs ----
      HFrag bk[4];
      #pragma unroll
      for (int c = 0; c < 4; ++c) {
        bk[c].u[0] = *(const uint4*)&Ks[buf][n][c * 32 + g * 16];
        bk[c].u[1] = *(const uint4*)&Ks[buf][n][c * 32 + g * 16 + 8];
      }
      v8f s0 = {};
      #pragma unroll
      for (int c = 0; c < 4; ++c)
        s0 = __builtin_amdgcn_wmma_f32_16x16x32_f16(
            false, a[c].h, false, bk[c].h, (short)0, s0, false, false);
      #pragma unroll
      for (int c = 0; c < 4; ++c) {
        bk[c].u[0] = *(const uint4*)&Ks[buf][16 + n][c * 32 + g * 16];
        bk[c].u[1] = *(const uint4*)&Ks[buf][16 + n][c * 32 + g * 16 + 8];
      }
      v8f s1 = {};
      #pragma unroll
      for (int c = 0; c < 4; ++c)
        s1 = __builtin_amdgcn_wmma_f32_16x16x32_f16(
            false, a[c].h, false, bk[c].h, (short)0, s1, false, false);

      const int rowb = q0 + 8 * g;

      // wave-uniform: only the diagonal tile needs the causal mask
      if (k0 + KTILE - 1 > q0) {
        #pragma unroll
        for (int r = 0; r < 8; ++r) {
          int rq = rowb + r;
          s0[r] += ((k0 + n)      > rq ? MASK2 : 0.0f);
          s1[r] += ((k0 + 16 + n) > rq ? MASK2 : 0.0f);
        }
      }

      // ---- packed-f16 row-max reduction (halves ds_bpermute count) ----
      // A 1-ulp-low f16 max is a valid softmax shift: used consistently,
      // p <= 2^0.008, still f16-representable.
      PKU pk[4];
      #pragma unroll
      for (int rp = 0; rp < 4; ++rp) {
        float m0 = fmaxf(s0[2 * rp],     s1[2 * rp]);
        float m1 = fmaxf(s0[2 * rp + 1], s1[2 * rp + 1]);
        auto t = __builtin_amdgcn_cvt_pkrtz(m0, m1);   // v_cvt_pk_rtz_f16_f32
        __builtin_memcpy(&pk[rp].h, &t, 4);
      }
      #pragma unroll
      for (int off = 1; off < 16; off <<= 1) {         // stays in 16-lane group
        #pragma unroll
        for (int rp = 0; rp < 4; ++rp) {
          PKU ot; ot.f = __shfl_xor(pk[rp].f, off, 32);
          pk[rp].h = __builtin_elementwise_max(pk[rp].h, ot.h);  // v_pk_max
        }
      }

      // ---- online update: alpha rescale + exp (row sums via WMMA below) ----
      float p0[8], p1[8];
      #pragma unroll
      for (int r = 0; r < 8; ++r) {
        float mtr = (float)pk[r >> 1].h[r & 1];
        float mn  = fmaxf(m_i[r], mtr);
        float al  = exp2f(m_i[r] - mn);          // native v_exp_f32
        p0[r] = exp2f(s0[r] - mn);
        p1[r] = exp2f(s1[r] - mn);
        m_i[r] = mn;
        ol[r] *= al;                             // denominator accumulator
        #pragma unroll
        for (int c = 0; c < 8; ++c) o[c][r] *= al;
      }

      // ---- P: C-layout -> per-wave LDS -> A-layout ----
      #pragma unroll
      for (int r = 0; r < 8; ++r) {
        Ps[wave][8 * g + r][n]      = (_Float16)p0[r];
        Ps[wave][8 * g + r][16 + n] = (_Float16)p1[r];
      }
      asm volatile("" ::: "memory");   // keep store->load order through Ps
      HFrag pa;
      pa.u[0] = *(const uint4*)&Ps[wave][n][g * 16];
      pa.u[1] = *(const uint4*)&Ps[wave][n][g * 16 + 8];

      // ---- denominator: ol += P * ones (row sums via matrix pipe) ----
      ol = __builtin_amdgcn_wmma_f32_16x16x32_f16(false, pa.h, false, ones,
                                                  (short)0, ol, false, false);

      // ---- O(16x128) += P(16x32) * V(32x128), 4-frag batches ----
      #pragma unroll
      for (int h = 0; h < 2; ++h) {
        HFrag vb[4];
        #pragma unroll
        for (int c = 0; c < 4; ++c) {
          vb[c].u[0] = *(const uint4*)&Vt[buf][(h * 4 + c) * 16 + n][g * 16];
          vb[c].u[1] = *(const uint4*)&Vt[buf][(h * 4 + c) * 16 + n][g * 16 + 8];
        }
        #pragma unroll
        for (int c = 0; c < 4; ++c)
          o[h * 4 + c] = __builtin_amdgcn_wmma_f32_16x16x32_f16(
              false, pa.h, false, vb[c].h, (short)0, o[h * 4 + c], false, false);
      }
    }

    // ---- publish next tile into the other buffer ----
    __syncthreads();                   // all waves done reading buf
    if (hasNext) storeKV(1 - buf, kd, vd);
    __syncthreads();                   // publish buffer 1-buf
  }

  // ---- epilogue: out[row][dim] = O / l  (l == P*ones accumulator) ----
  #pragma unroll
  for (int r = 0; r < 8; ++r) {
    float inv = 1.0f / ol[r];
    float* op = out + (bbase + q0 + 8 * g + r) * HEAD_DIM + n;
    #pragma unroll
    for (int c = 0; c < 8; ++c) op[c * 16] = o[c][r] * inv;
  }
}

// ---------------------------------------------------------------------------
extern "C" void kernel_launch(void* const* d_in, const int* in_sizes, int n_in,
                              void* d_out, int out_size, void* d_ws, size_t ws_size,
                              hipStream_t stream) {
  (void)n_in; (void)out_size; (void)ws_size;
  const float* qkv = (const float*)d_in[0];
  const int BS = in_sizes[0] / (3 * HEAD_DIM);   // B*S = 32768
  const int B  = BS / SEQ;

  size_t N = (size_t)BS * HEAD_DIM;              // halves per matrix
  _Float16* Qh = (_Float16*)d_ws;
  _Float16* Kh = Qh + N;
  _Float16* Vh = Kh + N;

  int total = BS * 192;                          // pairs across q,k,v
  rope_pack_kernel<<<(total + 255) / 256, 256, 0, stream>>>(qkv, Qh, Kh, Vh, BS);
  fattn_kernel<<<B * (SEQ / QTILE), 256, 0, stream>>>(Qh, Kh, Vh, (float*)d_out, SEQ);
}
